// ElectronicSchnet_76974403879066
// MI455X (gfx1250) — matrix operand order repaired and checked by reading, not compiled, not run on hardware
//
#include <hip/hip_runtime.h>
#include <hip/hip_bf16.h>

typedef __attribute__((ext_vector_type(16))) _Float16 v16h;
typedef __attribute__((ext_vector_type(8)))  _Float16 v8h;
typedef __attribute__((ext_vector_type(8)))  float    v8f;

#define N_ELEC 30
#define N_ALL  38
#define N_NUC  8
#define BASIS  32
#define KD     64
#define ED     128
#define HK     45      // round(sqrt(32*64))
#define HO     91      // round(sqrt(64*128))
#define HKP    48
#define HOP    96
#define NPAIR  1110    // 30 * 37
#define NTILE  70      // ceil(1110/16)

// ---- LDS layout (dynamic shared, ~132KB) ----
constexpr int OFF_XS   = 0;                        // f32 32x128
constexpr int OFF_MSG  = OFF_XS   + 32*128*4;      // f32 32x64
constexpr int OFF_ZS   = OFF_MSG  + 32*64*4;       // f32 38x64
constexpr int OFF_XSH  = OFF_ZS   + 38*64*4;       // f16 32x128
constexpr int OFF_MSGH = OFF_XSH  + 32*128*2;      // f16 32x64
constexpr int OFF_T    = OFF_MSGH + 32*64*2;       // f16 32x96
constexpr int OFF_W1   = OFF_T    + 32*96*2;       // f16 48x32  (N-major, N padded)
constexpr int OFF_W2   = OFF_W1   + 48*32*2;       // f16 64x64  (K padded 45->64)
constexpr int OFF_EW   = OFF_W2   + 64*64*2;       // f16 64x128
constexpr int OFF_O1   = OFF_EW   + 64*128*2;      // f16 96x64
constexpr int OFF_O2   = OFF_O1   + 96*64*2;       // f16 128x96 (K padded 91->96)
constexpr int OFF_B1   = OFF_O2   + 128*96*2;      // f32 48
constexpr int OFF_B2   = OFF_B1   + 48*4;          // f32 64
constexpr int OFF_OB1  = OFF_B2   + 64*4;          // f32 96
constexpr int OFF_OB2  = OFF_OB1  + 96*4;          // f32 128
constexpr int OFF_HST  = OFF_OB2  + 128*4;         // f16 8 waves x 16x64
constexpr int SMEM_BYTES = OFF_HST + 8*16*64*2;

__device__ __forceinline__ float ssp(float x) {
  float ax = fabsf(x);
  return fmaxf(x, 0.0f) + __logf(1.0f + __expf(-ax)) - 0.6931471805599453f;
}

__device__ __forceinline__ v8f wmma16(v16h a, v16h b, v8f c) {
  return __builtin_amdgcn_wmma_f32_16x16x32_f16(false, a, false, b, (short)0, c, false, false);
}

// A-fragment (16x32 f16) from row-major LDS matrix; ld in halves (multiple of 8)
__device__ __forceinline__ v16h ldsA(const _Float16* M, int row0, int ld, int k0, int lane) {
  const _Float16* p = M + (row0 + (lane & 15)) * ld + k0 + ((lane < 16) ? 0 : 8);
  v8h lo = *(const v8h*)p;
  v8h hi = *(const v8h*)(p + 16);
  v16h r;
#pragma unroll
  for (int t = 0; t < 8; ++t) { r[t] = lo[t]; r[t + 8] = hi[t]; }
  return r;
}

// B-fragment (32x16 f16) from N-major (transposed) LDS matrix; ld in halves (multiple of 16)
__device__ __forceinline__ v16h ldsB(const _Float16* Bt, int n0, int ld, int k0, int lane) {
  const _Float16* p = Bt + (n0 + (lane & 15)) * ld + k0 + ((lane < 16) ? 0 : 8);
  v8h lo = *(const v8h*)p;
  v8h hi = *(const v8h*)(p + 16);
  v16h r;
#pragma unroll
  for (int t = 0; t < 8; ++t) { r[t] = lo[t]; r[t + 8] = hi[t]; }
  return r;
}

// A-fragment of db gathered straight from global (pair rows), f32 -> f16
__device__ __forceinline__ v16h load_db_frag(const float* __restrict__ dists_b, int tile, int lane) {
  int r = tile * 16 + (lane & 15);
  int p = (r < NPAIR) ? r : 0;           // clamp ONLY the global read
  int i = p / 37;
  int jj = p - i * 37;
  int j = jj + ((jj >= i) ? 1 : 0);
  const float* row = dists_b + ((size_t)(i * N_ALL + j) << 5);
  int kb = (lane < 16) ? 0 : 8;
  float4 f0 = *(const float4*)(row + kb);
  float4 f1 = *(const float4*)(row + kb + 4);
  float4 f2 = *(const float4*)(row + kb + 16);
  float4 f3 = *(const float4*)(row + kb + 20);
  v16h a;
  a[0]  = (_Float16)f0.x; a[1]  = (_Float16)f0.y; a[2]  = (_Float16)f0.z; a[3]  = (_Float16)f0.w;
  a[4]  = (_Float16)f1.x; a[5]  = (_Float16)f1.y; a[6]  = (_Float16)f1.z; a[7]  = (_Float16)f1.w;
  a[8]  = (_Float16)f2.x; a[9]  = (_Float16)f2.y; a[10] = (_Float16)f2.z; a[11] = (_Float16)f2.w;
  a[12] = (_Float16)f3.x; a[13] = (_Float16)f3.y; a[14] = (_Float16)f3.z; a[15] = (_Float16)f3.w;
  return a;
}

__global__ __launch_bounds__(256)
void schnet_kernel(const float* __restrict__ dists, const float* __restrict__ emb_e,
                   const float* __restrict__ emb_n,
                   const float* __restrict__ w1, const float* __restrict__ b1,
                   const float* __restrict__ w2, const float* __restrict__ b2,
                   const float* __restrict__ ew,
                   const float* __restrict__ o1, const float* __restrict__ ob1,
                   const float* __restrict__ o2, const float* __restrict__ ob2,
                   float* __restrict__ out)
{
  extern __shared__ char smem[];
  float*    xs    = (float*)   (smem + OFF_XS);
  float*    msg   = (float*)   (smem + OFF_MSG);
  float*    zs    = (float*)   (smem + OFF_ZS);
  _Float16* xs_h  = (_Float16*)(smem + OFF_XSH);
  _Float16* msg_h = (_Float16*)(smem + OFF_MSGH);
  _Float16* tbuf  = (_Float16*)(smem + OFF_T);
  _Float16* W1t   = (_Float16*)(smem + OFF_W1);
  _Float16* W2t   = (_Float16*)(smem + OFF_W2);
  _Float16* EWt   = (_Float16*)(smem + OFF_EW);
  _Float16* O1t   = (_Float16*)(smem + OFF_O1);
  _Float16* O2t   = (_Float16*)(smem + OFF_O2);
  float*    b1v   = (float*)   (smem + OFF_B1);
  float*    b2v   = (float*)   (smem + OFF_B2);
  float*    ob1v  = (float*)   (smem + OFF_OB1);
  float*    ob2v  = (float*)   (smem + OFF_OB2);

  const int tid  = threadIdx.x;
  const int lane = tid & 31;
  const int wv   = tid >> 5;
  const int hi   = lane >> 4;
  const int l15  = lane & 15;
  const int b    = blockIdx.x;
  const float* dists_b = dists + (size_t)b * (N_ELEC * N_ALL * BASIS);
  _Float16* hstage = (_Float16*)(smem + OFF_HST) + wv * (16 * 64);

  // ---- init xs / xs_h (rows 30,31 zero), zero h-stage pad columns ----
  for (int idx = tid; idx < 32 * ED; idx += 256) {
    float v = (idx < N_ELEC * ED) ? emb_e[idx] : 0.0f;
    xs[idx] = v;
    xs_h[idx] = (_Float16)v;
  }
  for (int t2 = lane; t2 < 16 * 16; t2 += 32)
    hstage[(t2 >> 4) * 64 + 48 + (t2 & 15)] = (_Float16)0.0f;
  __syncthreads();

  for (int l = 0; l < 3; ++l) {
    // ---- stage weights: f16, transposed (N-major), zero-padded ----
    const float* w1l = w1 + l * BASIS * HK;
    const float* w2l = w2 + l * HK * KD;
    const float* ewl = ew + l * ED * KD;
    const float* o1l = o1 + l * KD * HO;
    const float* o2l = o2 + l * HO * ED;
    for (int idx = tid; idx < HKP * BASIS; idx += 256) {
      int n = idx >> 5, k = idx & 31;
      W1t[idx] = (_Float16)((n < HK) ? w1l[k * HK + n] : 0.0f);
    }
    for (int idx = tid; idx < KD * 64; idx += 256) {
      int n = idx >> 6, k = idx & 63;
      W2t[idx] = (_Float16)((k < HK) ? w2l[k * KD + n] : 0.0f);
    }
    for (int idx = tid; idx < KD * ED; idx += 256) {
      int n = idx >> 7, k = idx & 127;
      EWt[idx] = (_Float16)ewl[k * KD + n];
    }
    for (int idx = tid; idx < HOP * KD; idx += 256) {
      int n = idx >> 6, k = idx & 63;
      O1t[idx] = (_Float16)((n < HO) ? o1l[k * HO + n] : 0.0f);
    }
    for (int idx = tid; idx < ED * HOP; idx += 256) {
      int n = idx / HOP, k = idx % HOP;
      O2t[idx] = (_Float16)((k < HO) ? o2l[k * ED + n] : 0.0f);
    }
    for (int idx = tid; idx < HKP; idx += 256) b1v[idx]  = (idx < HK) ? b1[l * HK + idx] : 0.0f;
    for (int idx = tid; idx < KD;  idx += 256) b2v[idx]  = b2[l * KD + idx];
    for (int idx = tid; idx < HOP; idx += 256) ob1v[idx] = (idx < HO) ? ob1[l * HO + idx] : 0.0f;
    for (int idx = tid; idx < ED;  idx += 256) ob2v[idx] = ob2[l * ED + idx];
    __syncthreads();

    // ---- zs = [xs @ EW ; nuc], zero msg ----
    for (int idx = tid; idx < N_NUC * KD; idx += 256)
      zs[(N_ELEC + (idx >> 6)) * KD + (idx & 63)] = emb_n[idx];
    for (int idx = tid; idx < 32 * KD; idx += 256) msg[idx] = 0.0f;
    {
      int mt = wv >> 2, nt = wv & 3;   // 8 jobs, one per wave
      v8f acc = {};
#pragma unroll
      for (int kc = 0; kc < 4; ++kc) {
        v16h a  = ldsA(xs_h, mt * 16, ED, kc * 32, lane);
        v16h bb = ldsB(EWt, nt * 16, ED, kc * 32, lane);
        acc = wmma16(a, bb, acc);
      }
      int n = nt * 16 + l15;
#pragma unroll
      for (int v = 0; v < 8; ++v) {
        int m = mt * 16 + v + 8 * hi;
        if (m < N_ELEC) zs[m * KD + n] = acc[v];
      }
    }
    __syncthreads();

    // ---- hoist loop-invariant W1/W2 B-fragments + biases into registers ----
    v16h w1f[3];
#pragma unroll
    for (int nt = 0; nt < 3; ++nt) w1f[nt] = ldsB(W1t, nt * 16, BASIS, 0, lane);
    v16h w2f[4][2];
#pragma unroll
    for (int nt = 0; nt < 4; ++nt)
#pragma unroll
      for (int kc = 0; kc < 2; ++kc) w2f[nt][kc] = ldsB(W2t, nt * 16, 64, kc * 32, lane);
    float b1r[3], b2r[4];
#pragma unroll
    for (int nt = 0; nt < 3; ++nt) b1r[nt] = b1v[nt * 16 + l15];
#pragma unroll
    for (int nt = 0; nt < 4; ++nt) b2r[nt] = b2v[nt * 16 + l15];

    // ---- pair-tile loop: GEMM1 -> ssp -> GEMM2 -> fused gather-scatter sum ----
    // 2-stage software pipeline on the global db fragment
    v16h adb = load_db_frag(dists_b, wv, lane);
    for (int tile = wv; tile < NTILE; tile += 8) {
      v16h adbn = adb;
      if (tile + 8 < NTILE) adbn = load_db_frag(dists_b, tile + 8, lane);  // wave-uniform cond

      // GEMM1: h = ssp(db @ W1 + b1)   (K=32 == one WMMA per N-tile)
#pragma unroll
      for (int nt = 0; nt < 3; ++nt) {
        v8f acc = {};
        acc = wmma16(adb, w1f[nt], acc);
        int n = nt * 16 + l15;
#pragma unroll
        for (int v = 0; v < 8; ++v)
          hstage[(v + 8 * hi) * 64 + n] = (_Float16)ssp(acc[v] + b1r[nt]);
      }
      // GEMM2: Ws = h @ W2 + b2 (all 4 N-tiles first)
      v8f acc[4];
#pragma unroll
      for (int nt = 0; nt < 4; ++nt) {
        acc[nt] = (v8f){};
#pragma unroll
        for (int kc = 0; kc < 2; ++kc) {
          v16h a = ldsA(hstage, 0, 64, kc * 32, lane);
          acc[nt] = wmma16(a, w2f[nt][kc], acc[nt]);
        }
      }
      // branch-free scatter: pad rows (p>=1110) land in msg row 30 (discarded later)
#pragma unroll
      for (int v = 0; v < 8; ++v) {
        int p = tile * 16 + v + 8 * hi;          // < 1120
        int i = p / 37;                           // <= 30
        int jj = p - i * 37;
        int j = jj + ((jj >= i) ? 1 : 0);         // <= 37
        const float* zrow = zs + j * KD + l15;
        float* mrow = msg + i * KD + l15;
#pragma unroll
        for (int nt = 0; nt < 4; ++nt)
          atomicAdd(mrow + nt * 16, (acc[nt][v] + b2r[nt]) * zrow[nt * 16]);
      }
      adb = adbn;
    }
    __syncthreads();

    // ---- msg -> f16 ----
    for (int idx = tid; idx < 32 * KD; idx += 256) msg_h[idx] = (_Float16)msg[idx];
    __syncthreads();

    // ---- t = ssp(msg @ O1 + ob1) ----
    for (int job = wv; job < 12; job += 8) {
      int mt = job / 6, nt = job % 6;
      v8f acc = {};
#pragma unroll
      for (int kc = 0; kc < 2; ++kc) {
        v16h a  = ldsA(msg_h, mt * 16, KD, kc * 32, lane);
        v16h bb = ldsB(O1t, nt * 16, KD, kc * 32, lane);
        acc = wmma16(a, bb, acc);
      }
      int n = nt * 16 + l15;
      float bias = ob1v[n];
#pragma unroll
      for (int v = 0; v < 8; ++v) {
        int m = mt * 16 + v + 8 * hi;
        float val = (n < HO) ? ssp(acc[v] + bias) : 0.0f;  // zero pad cols 91..95
        tbuf[m * HOP + n] = (_Float16)val;
      }
    }
    __syncthreads();

    // ---- xs += t @ O2 + ob2 ----
    for (int job = wv; job < 16; job += 8) {
      int mt = job >> 3, nt = job & 7;
      v8f acc = {};
#pragma unroll
      for (int kc = 0; kc < 3; ++kc) {
        v16h a  = ldsA(tbuf, mt * 16, HOP, kc * 32, lane);
        v16h bb = ldsB(O2t, nt * 16, HOP, kc * 32, lane);
        acc = wmma16(a, bb, acc);
      }
      int n = nt * 16 + l15;
      float bias = ob2v[n];
#pragma unroll
      for (int v = 0; v < 8; ++v) {
        int m = mt * 16 + v + 8 * hi;
        if (m < N_ELEC) xs[m * ED + n] += acc[v] + bias;
      }
    }
    __syncthreads();

    // ---- refresh xs_h for next layer ----
    for (int idx = tid; idx < 32 * ED; idx += 256) xs_h[idx] = (_Float16)xs[idx];
    __syncthreads();
  }

  // ---- write out (B, 30, 128) f32 ----
  float* outb = out + (size_t)b * (N_ELEC * ED);
  for (int idx = tid; idx < N_ELEC * ED; idx += 256) outb[idx] = xs[idx];
}

extern "C" void kernel_launch(void* const* d_in, const int* in_sizes, int n_in,
                              void* d_out, int out_size, void* d_ws, size_t ws_size,
                              hipStream_t stream) {
  const float* dists = (const float*)d_in[0];
  const float* emb_e = (const float*)d_in[1];
  const float* emb_n = (const float*)d_in[2];
  const float* w1    = (const float*)d_in[3];
  const float* b1    = (const float*)d_in[4];
  const float* w2    = (const float*)d_in[5];
  const float* b2    = (const float*)d_in[6];
  const float* ew    = (const float*)d_in[7];
  const float* o1    = (const float*)d_in[8];
  const float* ob1   = (const float*)d_in[9];
  const float* o2    = (const float*)d_in[10];
  const float* ob2   = (const float*)d_in[11];
  float* out = (float*)d_out;

  int B = in_sizes[0] / (N_ELEC * N_ALL * BASIS);
  (void)hipFuncSetAttribute(reinterpret_cast<const void*>(schnet_kernel),
                            hipFuncAttributeMaxDynamicSharedMemorySize, SMEM_BYTES);
  schnet_kernel<<<dim3(B), dim3(256), SMEM_BYTES, stream>>>(
      dists, emb_e, emb_n, w1, b1, w2, b2, ew, o1, ob1, o2, ob2, out);
}
